// SimpleInteractionBlock_69114613727529
// MI455X (gfx1250) — matrix operand order repaired and checked by reading, not compiled, not run on hardware
//
#include <hip/hip_runtime.h>
#include <hip/hip_bf16.h>

// ---------------------------------------------------------------------------
// CDNA5 (gfx1250) wave32 WMMA bf16 implementation of the interaction block.
// D = A(16x32 bf16) * B(32x16 bf16) + C(16x16 f32) via v_wmma_f32_16x16x32_bf16
//
// Fragment layouts (ISA 7.12.2, wave32):
//   A 16x32 bf16 : lane lo = M; per-lane elements are two contiguous K-runs:
//                  k = 8*hi + [0,8)  and  k = 16 + 8*hi + [0,8)
//   B 32x16 bf16 : element e -> K = 8*hi + (e&7) + 16*(e>>3), N = lane&15
//   C/D 16x16 f32: element r -> M = r + 8*hi,  N = lane&15
//
// B operands are staged in LDS in *fragment-major* order so one fragment is a
// single contiguous 32-byte LDS read (2x ds_load_b128). A operands are read as
// b128 runs. GEMM epilogues are compile-time template variants (no branches).
// ---------------------------------------------------------------------------

typedef __attribute__((ext_vector_type(16))) __bf16 bf16x16;
typedef __attribute__((ext_vector_type(8)))  __bf16 bf16x8;
typedef __attribute__((ext_vector_type(8)))  float  f32x8;
typedef __attribute__((ext_vector_type(4)))  float  f32x4;

#define WMMA_BF16(a, b, c) \
    __builtin_amdgcn_wmma_f32_16x16x32_bf16(false, (a), false, (b), (short)0, (c), false, false)

// Epilogue modes (compile-time)
#define MODE_PLAIN   0  // out = g
#define MODE_SWISH   1  // out = swish(g)
#define MODE_ADDPRE  2  // out = g + add
#define MODE_RES     3  // out = swish(g) + add   (residual)

// Fast swish: g * v_rcp_f32(1 + v_exp_f32(-g)) -- avoids the IEEE-division
// Newton expansion (v_div_scale/v_div_fmas chain) in the epilogue.
__device__ __forceinline__ float swishf(float g) {
    return g * __builtin_amdgcn_rcpf(1.0f + __expf(-g));
}

// Build an A fragment from two contiguous 8-float runs in global memory.
__device__ __forceinline__ bf16x16 a_frag_from_global(const float* __restrict__ p0,
                                                      const float* __restrict__ p1)
{
    f32x4 r0a = *(const f32x4*)(p0);
    f32x4 r0b = *(const f32x4*)(p0 + 4);
    f32x4 r1a = *(const f32x4*)(p1);
    f32x4 r1b = *(const f32x4*)(p1 + 4);
    bf16x16 a;
    #pragma unroll
    for (int i = 0; i < 4; ++i) {
        a[i]      = (__bf16)r0a[i];
        a[i + 4]  = (__bf16)r0b[i];
        a[i + 8]  = (__bf16)r1a[i];
        a[i + 12] = (__bf16)r1b[i];
    }
    return a;
}

// Build an A fragment from two contiguous 8-bf16 runs in LDS.
__device__ __forceinline__ bf16x16 a_frag_from_lds(const __bf16* p0, const __bf16* p1)
{
    bf16x8 r0 = *(const bf16x8*)(p0);
    bf16x8 r1 = *(const bf16x8*)(p1);
    bf16x16 a;
    #pragma unroll
    for (int i = 0; i < 8; ++i) {
        a[i]     = r0[i];
        a[i + 8] = r1[i];
    }
    return a;
}

// ---------------------------------------------------------------------------
// Generic fused GEMM: out[n,128] = epilogue( concat(A,Acat)[n,32*KSTEPS] @ W^T + b )
// W is [128, 32*KSTEPS] (torch-style [out,in]). MODE selects the epilogue.
// Block = 256 thr = 8 waves; wave w owns rows [16*(8*bx+w), +16) x 128 cols.
// ---------------------------------------------------------------------------
template <int KSTEPS, int MODE>
__global__ __launch_bounds__(256) void gemm_bias_act_kernel(
    const float* __restrict__ A, const float* __restrict__ Acat,
    const float* __restrict__ W, const float* __restrict__ bias,
    const float* __restrict__ add, float* __restrict__ out,
    int nrows)
{
    constexpr int KDIM = 32 * KSTEPS;
    // Fragment-major W: sW[(((t*KSTEPS)+kk)*32 + lane)*16 + e]
    __shared__ __bf16 sW[8 * KSTEPS * 32 * 16];
    __shared__ float sBias[128];

    const int tid = threadIdx.x;
    for (int d = tid; d < 8 * KSTEPS * 512; d += 256) {
        int e = d & 15;
        int ln = (d >> 4) & 31;
        int rest = d >> 9;
        int kk = rest & (KSTEPS - 1);
        int t = rest / KSTEPS;
        int k = (kk << 5) + ((ln >> 4) << 3) + (e & 7) + ((e >> 3) << 4);
        int c = (t << 4) + (ln & 15);
        sW[d] = (__bf16)W[(size_t)c * KDIM + k];
    }
    if (tid < 128) sBias[tid] = bias ? bias[tid] : 0.0f;
    __syncthreads();

    const int wave = tid >> 5;
    const int lane = tid & 31;
    const int lo = lane & 15, hi = lane >> 4;
    const int tile = blockIdx.x * 8 + wave;
    const int ntiles = (nrows + 15) >> 4;
    if (tile >= ntiles) return;
    const size_t row0 = (size_t)tile << 4;

    // Preload all A fragments (fully unrolled -> stays in VGPRs).
    bf16x16 afr[KSTEPS];
    #pragma unroll
    for (int kk = 0; kk < KSTEPS; ++kk) {
        const float* base;
        int koff = kk << 5;
        if (KSTEPS == 8 && kk >= 4) { base = Acat; koff -= 128; }
        else                        { base = A; }
        const float* row = base + (row0 + lo) * 128 + koff + (hi << 3);
        afr[kk] = a_frag_from_global(row, row + 16);
    }

    for (int t = 0; t < 8; ++t) {           // 8 column tiles of 16
        f32x8 acc = {};
        #pragma unroll
        for (int kk = 0; kk < KSTEPS; ++kk) {
            bf16x16 b = *(const bf16x16*)&sW[(((t * KSTEPS) + kk) * 32 + lane) * 16];
            acc = WMMA_BF16(afr[kk], b, acc);
        }
        const int col = (t << 4) + lo;
        const float bv = sBias[col];
        #pragma unroll
        for (int r = 0; r < 8; ++r) {
            const size_t o = (row0 + r + (hi << 3)) * 128 + col;
            float g = acc[r] + bv;
            if (MODE == MODE_ADDPRE) g += add[o];
            if (MODE == MODE_SWISH || MODE == MODE_RES) g = swishf(g);
            if (MODE == MODE_RES) g += add[o];
            out[o] = g;
        }
    }
}

// ---------------------------------------------------------------------------
// Fused edge pipeline: f = (feat @ W1^T) @ W2^T ; msg = f * x[src] ;
// atomic scatter-add into agg[dst]. Wave handles a 16-edge tile.
// in_dim padded to 32*KS1 (KS1=2 for IN1=54, KS1=1 for IN2=18) with zeros.
// ---------------------------------------------------------------------------
template <int KS1>
__global__ __launch_bounds__(256) void edge_mlp_scatter_kernel(
    const float* __restrict__ feat, int in_dim,
    const float* __restrict__ W1,        // [64, in_dim]
    const float* __restrict__ W2,        // [128, 64]
    const float* __restrict__ xs,        // [N, 128] (swished node features)
    const int* __restrict__ src, const int* __restrict__ dst,
    float* __restrict__ agg, int E)
{
    __shared__ __bf16 sW1[4 * KS1 * 512];   // frag-major, zero-padded k>=in_dim
    __shared__ __bf16 sW2[8 * 2 * 512];     // frag-major (16 KB)
    __shared__ __bf16 sA[8][16 * 32 * KS1]; // per-wave padded feature tile, row-major
    __shared__ __bf16 sMid[8][16 * 64];     // per-wave mid activations, row-major
    __shared__ int sSrc[8][16];
    __shared__ int sDst[8][16];

    constexpr int KPAD = 32 * KS1;
    const int tid = threadIdx.x;

    for (int d = tid; d < 4 * KS1 * 512; d += 256) {
        int e = d & 15;
        int ln = (d >> 4) & 31;
        int rest = d >> 9;
        int kk = rest & (KS1 - 1);
        int mt = rest / KS1;
        int k = (kk << 5) + ((ln >> 4) << 3) + (e & 7) + ((e >> 3) << 4);
        int m = (mt << 4) + (ln & 15);
        sW1[d] = (k < in_dim) ? (__bf16)W1[(size_t)m * in_dim + k] : (__bf16)0.0f;
    }
    for (int d = tid; d < 8 * 2 * 512; d += 256) {
        int e = d & 15;
        int ln = (d >> 4) & 31;
        int rest = d >> 9;
        int kk = rest & 1;
        int t = rest >> 1;
        int k = (kk << 5) + ((ln >> 4) << 3) + (e & 7) + ((e >> 3) << 4);
        int o = (t << 4) + (ln & 15);
        sW2[d] = (__bf16)W2[(size_t)o * 64 + k];
    }

    const int wave = tid >> 5;
    const int lane = tid & 31;
    const int e0 = (blockIdx.x * 8 + wave) * 16;
    const bool active = (e0 < E);
    if (active) {
        for (int i = lane; i < 16 * KPAD; i += 32) {
            int m = i / KPAD, k = i % KPAD;
            sA[wave][i] = (k < in_dim)
                              ? (__bf16)feat[(size_t)(e0 + m) * in_dim + k]
                              : (__bf16)0.0f;
        }
        if (lane < 16) {
            sSrc[wave][lane] = src[e0 + lane];
            sDst[wave][lane] = dst[e0 + lane];
        }
    }
    __syncthreads();
    if (!active) return;

    const int lo = lane & 15, hi = lane >> 4;

    // ---- GEMM 1: mid(16x64) = A(16xKPAD) @ W1^T ----
    bf16x16 afr[KS1];
    #pragma unroll
    for (int kk = 0; kk < KS1; ++kk) {
        const __bf16* row = &sA[wave][lo * KPAD + (kk << 5) + (hi << 3)];
        afr[kk] = a_frag_from_lds(row, row + 16);
    }
    #pragma unroll
    for (int mt = 0; mt < 4; ++mt) {
        f32x8 acc = {};
        #pragma unroll
        for (int kk = 0; kk < KS1; ++kk) {
            bf16x16 b = *(const bf16x16*)&sW1[(((mt * KS1) + kk) * 32 + lane) * 16];
            acc = WMMA_BF16(afr[kk], b, acc);
        }
        #pragma unroll
        for (int r = 0; r < 8; ++r)
            sMid[wave][(r + (hi << 3)) * 64 + (mt << 4) + lo] = (__bf16)acc[r];
    }

    // ---- GEMM 2: f(16x128) = mid(16x64) @ W2^T ; multiply & scatter ----
    bf16x16 mfr[2];
    #pragma unroll
    for (int kk = 0; kk < 2; ++kk) {
        const __bf16* row = &sMid[wave][lo * 64 + (kk << 5) + (hi << 3)];
        mfr[kk] = a_frag_from_lds(row, row + 16);
    }
    for (int t = 0; t < 8; ++t) {
        f32x8 acc = {};
        #pragma unroll
        for (int kk = 0; kk < 2; ++kk) {
            bf16x16 b = *(const bf16x16*)&sW2[(((t * 2) + kk) * 32 + lane) * 16];
            acc = WMMA_BF16(mfr[kk], b, acc);
        }
        const int col = (t << 4) + lo;
        #pragma unroll
        for (int r = 0; r < 8; ++r) {
            int m = r + (hi << 3);
            float v = acc[r] * xs[(size_t)sSrc[wave][m] * 128 + col];
            atomicAdd(&agg[(size_t)sDst[wave][m] * 128 + col], v);
        }
    }
}

// ---------------------------------------------------------------------------
// GraphNorm (batch is sorted, B=512 segments)
// ---------------------------------------------------------------------------
__global__ void gn_pass1(const float* __restrict__ h, const int* __restrict__ batch,
                         float* __restrict__ sums, float* __restrict__ cnt, int N)
{
    int idx = blockIdx.x * blockDim.x + threadIdx.x;
    if (idx >= N * 128) return;
    int n = idx >> 7, c = idx & 127;
    int b = batch[n];
    atomicAdd(&sums[b * 128 + c], h[idx]);
    if (c == 0) atomicAdd(&cnt[b], 1.0f);
}

__global__ void gn_div(float* __restrict__ v, const float* __restrict__ cnt, int B)
{
    int idx = blockIdx.x * blockDim.x + threadIdx.x;
    if (idx >= B * 128) return;
    float c = cnt[idx >> 7];
    v[idx] *= __builtin_amdgcn_rcpf(fmaxf(c, 1.0f));
}

__global__ void gn_pass2(const float* __restrict__ h, const int* __restrict__ batch,
                         const float* __restrict__ mean, const float* __restrict__ ms,
                         float* __restrict__ var, int N)
{
    int idx = blockIdx.x * blockDim.x + threadIdx.x;
    if (idx >= N * 128) return;
    int n = idx >> 7, c = idx & 127;
    int b = batch[n];
    float hc = h[idx] - mean[b * 128 + c] * ms[c];
    atomicAdd(&var[b * 128 + c], hc * hc);
}

__global__ void gn_apply(float* __restrict__ h, const int* __restrict__ batch,
                         const float* __restrict__ mean, const float* __restrict__ var,
                         const float* __restrict__ ms, const float* __restrict__ w,
                         const float* __restrict__ bb, int N)
{
    int idx = blockIdx.x * blockDim.x + threadIdx.x;
    if (idx >= N * 128) return;
    int n = idx >> 7, c = idx & 127;
    int b = batch[n];
    float hc = h[idx] - mean[b * 128 + c] * ms[c];
    h[idx] = w[c] * hc * rsqrtf(var[b * 128 + c] + 1e-5f) + bb[c];
}

// ---------------------------------------------------------------------------
extern "C" void kernel_launch(void* const* d_in, const int* in_sizes, int n_in,
                              void* d_out, int out_size, void* d_ws, size_t ws_size,
                              hipStream_t stream)
{
    const int H = 128, B = 512;
    const int N = in_sizes[0] / H;
    const int E = in_sizes[1] / 54;

    const float* x        = (const float*)d_in[0];
    const float* feature1 = (const float*)d_in[1];
    const float* feature2 = (const float*)d_in[2];
    const int*   edge     = (const int*)  d_in[3];
    const int*   batch    = (const int*)  d_in[4];
    const float* lin_W    = (const float*)d_in[5];
    const float* lin_b    = (const float*)d_in[6];
    const float* f1_W1    = (const float*)d_in[7];
    const float* f1_W2    = (const float*)d_in[8];
    const float* f2_W1    = (const float*)d_in[9];
    const float* f2_W2    = (const float*)d_in[10];
    const float* rel1_W   = (const float*)d_in[11];
    const float* rel1_b   = (const float*)d_in[12];
    const float* root1_W  = (const float*)d_in[13];
    const float* rel2_W   = (const float*)d_in[14];
    const float* rel2_b   = (const float*)d_in[15];
    const float* root2_W  = (const float*)d_in[16];
    const float* lin1_W   = (const float*)d_in[17];
    const float* lin1_b   = (const float*)d_in[18];
    const float* lin2_W   = (const float*)d_in[19];
    const float* lin2_b   = (const float*)d_in[20];
    const float* cat_W    = (const float*)d_in[21];
    const float* cat_b    = (const float*)d_in[22];
    const float* norm_w   = (const float*)d_in[23];
    const float* norm_b   = (const float*)d_in[24];
    const float* norm_ms  = (const float*)d_in[25];
    const float* lins_W   = (const float*)d_in[26];
    const float* lins_b   = (const float*)d_in[27];
    const float* final_W  = (const float*)d_in[28];
    const float* final_b  = (const float*)d_in[29];

    const int* srcp = edge;
    const int* dstp = edge + E;

    char* ws = (char*)d_ws;
    const size_t nh = (size_t)N * H * sizeof(float);
    float* xs   = (float*)(ws + 0 * nh);   // swish(x @ lin)
    float* b1   = (float*)(ws + 1 * nh);   // agg1 -> h1s
    float* b2   = (float*)(ws + 2 * nh);   // agg2 -> h2s
    float* b3   = (float*)(ws + 3 * nh);   // h1 combine -> h (ping)
    float* b4   = (float*)(ws + 4 * nh);   // h2 combine -> h (pong)
    float* sums = (float*)(ws + 5 * nh);   // [B,128] mean
    float* var  = sums + (size_t)B * H;    // [B,128]
    float* cnt  = var + (size_t)B * H;     // [B]

    const int gtiles = (N + 15) / 16;
    const int ggrid = (gtiles + 7) / 8;
    const int egrid = (E / 16 + 7) / 8;
    const int eN = (N * H + 255) / 256;
    const int eB = (B * H + 255) / 256;

    #define GEMM(MODE_, A_, W_, bias_, add_, out_) \
        gemm_bias_act_kernel<4, MODE_><<<ggrid, 256, 0, stream>>>(A_, nullptr, W_, bias_, add_, out_, N)

    // x_s = swish(x @ lin_W^T + lin_b)
    GEMM(MODE_SWISH, x, lin_W, lin_b, nullptr, xs);

    // zero scatter targets + stats
    hipMemsetAsync(b1, 0, nh, stream);
    hipMemsetAsync(b2, 0, nh, stream);
    hipMemsetAsync(sums, 0, (size_t)(2 * B * H + B) * sizeof(float), stream);

    // fused edge pipelines -> agg1 (b1), agg2 (b2)
    edge_mlp_scatter_kernel<2><<<egrid, 256, 0, stream>>>(feature1, 54, f1_W1, f1_W2,
                                                          xs, srcp, dstp, b1, E);
    edge_mlp_scatter_kernel<1><<<egrid, 256, 0, stream>>>(feature2, 18, f2_W1, f2_W2,
                                                          xs, srcp, dstp, b2, E);

    // conv1 combine + lin1
    GEMM(MODE_PLAIN,  b1, rel1_W, rel1_b, nullptr, b3);   // agg1 @ rel + b
    GEMM(MODE_ADDPRE, xs, root1_W, nullptr, b3, b3);      // += x @ root
    GEMM(MODE_SWISH,  b3, lin1_W, lin1_b, nullptr, b1);   // h1s = swish(.)

    // conv2 combine + lin2
    GEMM(MODE_PLAIN,  b2, rel2_W, rel2_b, nullptr, b4);
    GEMM(MODE_ADDPRE, xs, root2_W, nullptr, b4, b4);
    GEMM(MODE_SWISH,  b4, lin2_W, lin2_b, nullptr, b2);   // h2s

    // h = concat(h1s, h2s) @ cat_W^T + cat_b + x_s
    gemm_bias_act_kernel<8, MODE_ADDPRE><<<ggrid, 256, 0, stream>>>(b1, b2, cat_W, cat_b, xs, b3, N);

    // residual stack: h = swish(h @ W + b) + h  (ping-pong b3 <-> b4)
    GEMM(MODE_RES, b3, lins_W + 0 * H * H, lins_b + 0 * H, b3, b4);
    GEMM(MODE_RES, b4, lins_W + 1 * H * H, lins_b + 1 * H, b4, b3);
    GEMM(MODE_RES, b3, lins_W + 2 * H * H, lins_b + 2 * H, b3, b4);

    // GraphNorm in place on b4
    gn_pass1<<<eN, 256, 0, stream>>>(b4, batch, sums, cnt, N);
    gn_div<<<eB, 256, 0, stream>>>(sums, cnt, B);
    gn_pass2<<<eN, 256, 0, stream>>>(b4, batch, sums, norm_ms, var, N);
    gn_div<<<eB, 256, 0, stream>>>(var, cnt, B);
    gn_apply<<<eN, 256, 0, stream>>>(b4, batch, sums, var, norm_ms, norm_w, norm_b, N);

    // out = h @ final_W^T + final_b
    GEMM(MODE_PLAIN, b4, final_W, final_b, nullptr, (float*)d_out);

    #undef GEMM
}
